// SpikeFP32Softmax_29875792511515
// MI455X (gfx1250) — matrix operand order repaired and checked by reading, not compiled, not run on hardware
//
#include <hip/hip_runtime.h>
#include <stdint.h>

// SpikeFP32 softmax: [8,1024,256,32] spike-encoded FP32 -> softmax over N=256 -> spikes.
// Memory-bound (512 MB @ 23.3 TB/s ~ 22us). Bit-decode done exactly with
// v_wmma_f32_16x16x32_bf16: hi/lo 16-bit halves each fit exactly in f32
// (<=16 distinct powers of two spanning 16 octaves < 24-bit mantissa).

typedef __attribute__((ext_vector_type(16))) __bf16 bf16x16;
typedef __attribute__((ext_vector_type(8)))  float  f32x8;

union ABits {
    uint32_t u[8];
    bf16x16  v;
};

__device__ __forceinline__ uint32_t pack_bf16x2(float lo, float hi) {
    // bf16(lo) in [15:0], bf16(hi) in [31:16]; exact for 0.0f/1.0f.
    // One v_perm_b32: D = {hi.b3, hi.b2, lo.b3, lo.b2}
    return __builtin_amdgcn_perm(__float_as_uint(hi), __float_as_uint(lo), 0x07060302u);
}

__global__ __launch_bounds__(256) void spike_fp32_softmax_kernel(
    const float* __restrict__ x, float* __restrict__ out, int rows)
{
    const int row = blockIdx.x;
    if (row >= rows) return;

    const int t    = threadIdx.x;     // 0..255, one element of the row
    const int lane = t & 31;
    const int wave = t >> 5;          // 0..7, 32 elements per wave
    const size_t rowBase = (size_t)row * 256u * 32u;

    __shared__ float s_hilo[8][2][32];   // per wave: [hi|lo][elem-in-wave]
    __shared__ float s_part[8];

    // ---- B matrix (32x16 bf16): col0 = hi weights 2^(31-k), k=0..15
    //                             col1 = lo weights 2^(31-k), k=16..31
    // Layout: lanes 0-15 hold K=0..15 of column `lane` (VGPR j = K pair 2j,2j+1),
    //         lanes 16-31 hold K=16..31 of column `lane-16`.
    ABits b;
#pragma unroll
    for (int i = 0; i < 8; ++i) b.u[i] = 0u;
    if (lane == 0) {
#pragma unroll
        for (int i = 0; i < 16; ++i) {
            uint32_t bf = (uint32_t)(127 + (31 - i)) << 7;   // bf16 bits of 2^(31-i)
            b.u[i >> 1] |= bf << ((i & 1) * 16);
        }
    } else if (lane == 17) {
#pragma unroll
        for (int i = 0; i < 16; ++i) {
            uint32_t bf = (uint32_t)(127 + (15 - i)) << 7;   // bf16 bits of 2^(15-i)
            b.u[i >> 1] |= bf << ((i & 1) * 16);
        }
    }

    // ---- Decode 32 elements per wave with two WMMAs.
    // A 16x32 as loaded here (same K order as B): lanes 0-15 (row M=lane) hold K=0..15 (V0-7);
    //                        lanes 16-31 (row M=lane-16) hold K=16..31.
    const int mrow = lane & 15;
    const int halfSel = lane >> 4;               // 0: bits 0..15, 1: bits 16..31
    const int Ncol  = mrow;                      // D-tile column this lane holds
    const int Mbase = (lane < 16) ? 0 : 8;

#pragma unroll
    for (int tt = 0; tt < 2; ++tt) {
        const int elem = wave * 32 + tt * 16 + mrow;
        const float4* p = (const float4*)(x + rowBase + (size_t)elem * 32u + (size_t)halfSel * 16u);
        float4 f0 = p[0];   // bits off+0..3   (off = halfSel*16)
        float4 f1 = p[1];   // bits off+4..7
        float4 f2 = p[2];   // bits off+8..11
        float4 f3 = p[3];   // bits off+12..15

        ABits a;
        a.u[0] = pack_bf16x2(f0.x, f0.y);  a.u[1] = pack_bf16x2(f0.z, f0.w);
        a.u[2] = pack_bf16x2(f1.x, f1.y);  a.u[3] = pack_bf16x2(f1.z, f1.w);
        a.u[4] = pack_bf16x2(f2.x, f2.y);  a.u[5] = pack_bf16x2(f2.z, f2.w);
        a.u[6] = pack_bf16x2(f3.x, f3.y);  a.u[7] = pack_bf16x2(f3.z, f3.w);

        f32x8 c = {};
        c = __builtin_amdgcn_wmma_f32_16x16x32_bf16(
                false, a.v, false, b.v, (short)0, c, false, false);

        // D[M][N]: vgpr r -> M = Mbase + r, N = Ncol. Only columns 0 (hi) and 1 (lo) matter.
        if (Ncol < 2) {
#pragma unroll
            for (int r = 0; r < 8; ++r)
                s_hilo[wave][Ncol][tt * 16 + Mbase + r] = c[r];
        }
    }
    __syncthreads();

    // ---- Reassemble IEEE word (exact: hi,lo are exact f32 integers), exp.
    const uint32_t uhi = (uint32_t)s_hilo[wave][0][lane];
    const uint32_t ulo = (uint32_t)s_hilo[wave][1][lane];
    const float xv = __uint_as_float(uhi | ulo);
    const float ex = __expf(xv);

    // ---- Row sum: intra-wave shfl_xor tree + cross-wave LDS combine.
    float s = ex;
#pragma unroll
    for (int o = 16; o > 0; o >>= 1) s += __shfl_xor(s, o, 32);
    if (lane == 0) s_part[wave] = s;
    __syncthreads();
    float total = 0.0f;
#pragma unroll
    for (int w = 0; w < 8; ++w) total += s_part[w];

    // ---- Divide and re-encode to 32 spike floats (MSB first), 128B/thread.
    const uint32_t u = __float_as_uint(ex / total);
    float4* q = (float4*)(out + rowBase + (size_t)t * 32u);
#pragma unroll
    for (int g = 0; g < 8; ++g) {
        float4 v;
        v.x = (float)((u >> (31 - (g * 4 + 0))) & 1u);
        v.y = (float)((u >> (31 - (g * 4 + 1))) & 1u);
        v.z = (float)((u >> (31 - (g * 4 + 2))) & 1u);
        v.w = (float)((u >> (31 - (g * 4 + 3))) & 1u);
        q[g] = v;
    }
}

extern "C" void kernel_launch(void* const* d_in, const int* in_sizes, int n_in,
                              void* d_out, int out_size, void* d_ws, size_t ws_size,
                              hipStream_t stream) {
    const float* x = (const float*)d_in[0];
    float* out = (float*)d_out;
    const int rows = in_sizes[0] / (256 * 32);   // 8*1024 = 8192 rows of N=256
    dim3 grid((unsigned)rows), block(256);
    spike_fp32_softmax_kernel<<<grid, block, 0, stream>>>(x, out, rows);
}
